// _PointnetSAModuleBase_876173328634
// MI455X (gfx1250) — matrix operand order repaired and checked by reading, not compiled, not run on hardware
//
#include <hip/hip_runtime.h>
#include <hip/hip_bf16.h>

#define BATCH   8
#define NPTS    16384
#define MCTR    1024
#define NSAMP   64
#define RAD2    0.64f

typedef __attribute__((ext_vector_type(16))) _Float16 v16h;
typedef __attribute__((ext_vector_type(8)))  _Float16 v8h;
typedef __attribute__((ext_vector_type(8)))  float    v8f;

union V16 { v16h v; v8h h[2]; };

__device__ __forceinline__ v8f wmma32(v16h a, v16h b, v8f c) {
  return __builtin_amdgcn_wmma_f32_16x16x32_f16(false, a, false, b, (short)0, c, false, false);
}

// A tile: 16x32 f16 weights, row-major in LDS with stride WS halves.
// lane L: row = base_row + (L&15); kb = ks*32 + 16*(L>>4); contiguous 16 halves [kb,kb+16)
__device__ __forceinline__ v16h load_a(const _Float16* W, int WS, int row, int kbase) {
  V16 a;
  const _Float16* p = W + row * WS + kbase;
  a.h[0] = *(const v8h*)(p);
  a.h[1] = *(const v8h*)(p + 8);
  return a.v;
}

// B tile: 32x16 f16 activations, column-major in LDS with stride XS halves.
// lane L: col = base_col + (L&15); k0 = ks*32 + 16*(L>>4); contiguous 16 halves
__device__ __forceinline__ v16h load_b(const _Float16* X, int XS, int col, int k0) {
  V16 b;
  const _Float16* p = X + col * XS + k0;
  b.h[0] = *(const v8h*)(p);
  b.h[1] = *(const v8h*)(p + 8);
  return b.v;
}

// One MLP layer: Out(16*otiles x 64) = ReLU(W @ Xin + bias), written f16 column-major.
__device__ __forceinline__ void run_layer(const _Float16* W, int WS, const float* bias,
                                          const _Float16* Xin, int XSin, int ksteps,
                                          _Float16* Xout, int XSout, int otiles, int lane) {
  const int lm = lane & 15, lh = lane >> 4;
  for (int ot = 0; ot < otiles; ++ot) {
    const int row = ot * 16 + lm;
    for (int nt = 0; nt < 4; ++nt) {
      const int col = nt * 16 + lm;
      v8f acc = {0.f, 0.f, 0.f, 0.f, 0.f, 0.f, 0.f, 0.f};
      for (int ks = 0; ks < ksteps; ++ks) {
        v16h a  = load_a(W, WS, row, ks * 32 + lh * 16);
        v16h bm = load_b(Xin, XSin, col, ks * 32 + lh * 16);
        acc = wmma32(a, bm, acc);
      }
      const int rb = ot * 16 + lh * 8;
      v8h o;
#pragma unroll
      for (int v = 0; v < 8; ++v)
        o[v] = (_Float16)fmaxf(acc[v] + bias[rb + v], 0.f);
      *(v8h*)(Xout + col * XSout + rb) = o;
    }
  }
}

// ---------------- Kernel 1: furthest point sampling ----------------
// One block per batch, 1024 threads, points + dist resident in LDS.
__global__ void fps_kernel(const float* __restrict__ xyz, float* __restrict__ newxyz) {
  extern __shared__ char smem[];
  float* sx = (float*)smem;
  float* sy = sx + NPTS;
  float* sz = sy + NPTS;
  float* sd = sz + NPTS;
  float* rv = sd + NPTS;               // 32
  int*   ri = (int*)(rv + 32);         // 32
  int*   sfar = ri + 32;               // 1

  const int b = blockIdx.x, tid = threadIdx.x;
  const float* base = xyz + (size_t)b * NPTS * 3;
  for (int i = tid; i < NPTS; i += 1024) {
    sx[i] = base[i * 3 + 0];
    sy[i] = base[i * 3 + 1];
    sz[i] = base[i * 3 + 2];
    sd[i] = 1e10f;
  }
  __syncthreads();

  int far = 0;
  for (int s = 0; s < MCTR; ++s) {
    const float px = sx[far], py = sy[far], pz = sz[far];
    if (tid == 0) {
      float* o = newxyz + ((size_t)b * MCTR + s) * 3;
      o[0] = px; o[1] = py; o[2] = pz;
    }
    float bestv = -1.f; int besti = 0;
#pragma unroll 4
    for (int t = 0; t < 16; ++t) {
      const int i = tid + t * 1024;
      const float dx = sx[i] - px, dy = sy[i] - py, dz = sz[i] - pz;
      const float d = dx * dx + dy * dy + dz * dz;
      const float nd = fminf(sd[i], d);
      sd[i] = nd;
      if (nd > bestv) { bestv = nd; besti = i; }
    }
    for (int off = 16; off; off >>= 1) {
      const float ov = __shfl_down(bestv, off, 32);
      const int   oi = __shfl_down(besti, off, 32);
      if (ov > bestv || (ov == bestv && oi < besti)) { bestv = ov; besti = oi; }
    }
    if ((tid & 31) == 0) { rv[tid >> 5] = bestv; ri[tid >> 5] = besti; }
    __syncthreads();
    if (tid < 32) {
      bestv = rv[tid]; besti = ri[tid];
      for (int off = 16; off; off >>= 1) {
        const float ov = __shfl_down(bestv, off, 32);
        const int   oi = __shfl_down(besti, off, 32);
        if (ov > bestv || (ov == bestv && oi < besti)) { bestv = ov; besti = oi; }
      }
      if (tid == 0) *sfar = besti;
    }
    __syncthreads();
    far = *sfar;
  }
}

// ---------------- Kernel 2: ball query ----------------
// One wave per center: ordered first-64 selection via ballot + prefix popcount.
__global__ void ballq_kernel(const float* __restrict__ xyz, const float* __restrict__ newxyz,
                             int* __restrict__ gidx, int* __restrict__ gcnt) {
  const int wid  = (int)((blockIdx.x * blockDim.x + threadIdx.x) >> 5);
  const int lane = threadIdx.x & 31;
  const int b = wid >> 10;
  const float* base = xyz + (size_t)b * NPTS * 3;
  const float cx = newxyz[wid * 3 + 0];
  const float cy = newxyz[wid * 3 + 1];
  const float cz = newxyz[wid * 3 + 2];
  int* out = gidx + (size_t)wid * NSAMP;

  int cnt = 0;
  for (int i0 = 0; i0 < NPTS && cnt < NSAMP; i0 += 32) {
    const int i = i0 + lane;
    // pull the chunk after next into cache while we process this one
    __builtin_prefetch(base + (i + 64) * 3, 0, 1);
    const float dx = base[i * 3 + 0] - cx;
    const float dy = base[i * 3 + 1] - cy;
    const float dz = base[i * 3 + 2] - cz;
    const bool in = (dx * dx + dy * dy + dz * dz) < RAD2;
    const unsigned bal = (unsigned)__ballot(in);
    const int rank = __popc(bal & ((1u << lane) - 1u));
    if (in && (cnt + rank) < NSAMP) out[cnt + rank] = i;
    cnt += __popc(bal);
  }
  cnt = cnt < NSAMP ? cnt : NSAMP;
  const int first = (cnt > 0) ? out[0] : 0;
  for (int j = cnt + lane; j < NSAMP; j += 32) out[j] = first;
  if (lane == 0) gcnt[wid] = cnt;
}

// ---------------- Kernel 3: gather + 3-layer WMMA MLP + masked max-pool ----------------
// 128 threads = 4 waves per block; one wave per center.
// Per-wave LDS: X1 (96x64, stride 104) + Xa (64x64, stride 72); layer-2 output Xb
// ALIASES X1 (dead after layer 1) so two blocks fit per 320KB WGP (2 waves/SIMD).
#define W1S 104   // K1 padded 96 (+8 anti-conflict)
#define W2S 72    // K 64 (+8)
#define PERWAVE_H (64*104 + 64*72)   // X1(+Xb alias), Xa (halves)

__global__ void mlp_kernel(const float* __restrict__ xyz, const float* __restrict__ features,
                           const float* __restrict__ W1, const float* __restrict__ b1,
                           const float* __restrict__ W2, const float* __restrict__ b2,
                           const float* __restrict__ W3, const float* __restrict__ b3,
                           const float* __restrict__ newxyz,
                           const int* __restrict__ gidx, const int* __restrict__ gcnt,
                           float* __restrict__ outfeat) {
  extern __shared__ char smem[];
  _Float16* w1 = (_Float16*)smem;          // 64*104
  _Float16* w2 = w1 + 64 * W1S;            // 64*72
  _Float16* w3 = w2 + 64 * W2S;            // 128*72
  float*    bb = (float*)(w3 + 128 * W2S); // b1[64] b2[64] b3[128]
  _Float16* xbase = (_Float16*)(bb + 256);

  const int tid = threadIdx.x;
  // convert weights f32 -> f16 into padded LDS (zero-fill pads so garbage K never pollutes)
  for (int i = tid; i < 64 * W1S; i += 128) {
    const int o = i / W1S, k = i - o * W1S;
    w1[i] = (k < 67) ? (_Float16)W1[o * 67 + k] : (_Float16)0.f;
  }
  for (int i = tid; i < 64 * W2S; i += 128) {
    const int o = i / W2S, k = i - o * W2S;
    w2[i] = (k < 64) ? (_Float16)W2[o * 64 + k] : (_Float16)0.f;
  }
  for (int i = tid; i < 128 * W2S; i += 128) {
    const int o = i / W2S, k = i - o * W2S;
    w3[i] = (k < 64) ? (_Float16)W3[o * 64 + k] : (_Float16)0.f;
  }
  for (int i = tid; i < 64; i += 128)  bb[i] = b1[i];
  for (int i = tid; i < 64; i += 128)  bb[64 + i] = b2[i];
  for (int i = tid; i < 128; i += 128) bb[128 + i] = b3[i];
  __syncthreads();

  const int w = tid >> 5, lane = tid & 31;
  const int wid = blockIdx.x * 4 + w;            // center id
  const int b = wid >> 10, m = wid & 1023;
  _Float16* X1 = xbase + (size_t)w * PERWAVE_H;  // 96x64 col-major stride 104
  _Float16* Xa = X1 + 64 * W1S;                  // 64x64 col-major stride 72
  _Float16* Xb = X1;                             // layer-2 out aliases X1 (stride 72)

  // ---- gather: rows 0-2 = centered xyz, 3-66 = features, 67-95 = 0 ----
  const int* idx = gidx + (size_t)wid * NSAMP;
  const float cx = newxyz[wid * 3 + 0];
  const float cy = newxyz[wid * 3 + 1];
  const float cz = newxyz[wid * 3 + 2];
  const float* fb = features + (size_t)b * 64 * NPTS;
  const float* pb = xyz + (size_t)b * NPTS * 3;
#pragma unroll
  for (int jj = 0; jj < 2; ++jj) {
    const int j = lane + jj * 32;
    const int ij = idx[j];
    _Float16* col = X1 + j * W1S;
    col[0] = (_Float16)(pb[ij * 3 + 0] - cx);
    col[1] = (_Float16)(pb[ij * 3 + 1] - cy);
    col[2] = (_Float16)(pb[ij * 3 + 2] - cz);
    for (int c = 0; c < 64; ++c) col[3 + c] = (_Float16)fb[(size_t)c * NPTS + ij];
    for (int k = 67; k < 96; ++k) col[k] = (_Float16)0.f;
  }
  asm volatile("s_wait_dscnt 0" ::: "memory");

  run_layer(w1, W1S, bb,      X1, W1S, 3, Xa, W2S, 4, lane);
  asm volatile("s_wait_dscnt 0" ::: "memory");
  run_layer(w2, W2S, bb + 64, Xa, W2S, 2, Xb, W2S, 4, lane);
  asm volatile("s_wait_dscnt 0" ::: "memory");

  // ---- layer 3 (128 outputs) fused with masked max-pool over 64 samples ----
  const float scale = (gcnt[wid] > 0) ? 1.f : 0.f;
  const int lm = lane & 15, lh = lane >> 4;
  for (int ot = 0; ot < 8; ++ot) {
    float vmax[8];
#pragma unroll
    for (int v = 0; v < 8; ++v) vmax[v] = 0.f;
    for (int nt = 0; nt < 4; ++nt) {
      v8f acc = {0.f, 0.f, 0.f, 0.f, 0.f, 0.f, 0.f, 0.f};
      for (int ks = 0; ks < 2; ++ks) {
        v16h a  = load_a(w3, W2S, ot * 16 + lm, ks * 32 + lh * 16);
        v16h bm = load_b(Xb, W2S, nt * 16 + lm, ks * 32 + lh * 16);
        acc = wmma32(a, bm, acc);
      }
      const int rb = ot * 16 + lh * 8;
#pragma unroll
      for (int v = 0; v < 8; ++v)
        vmax[v] = fmaxf(vmax[v], fmaxf(acc[v] + bb[128 + rb + v], 0.f));
    }
#pragma unroll
    for (int v = 0; v < 8; ++v) {
      vmax[v] = fmaxf(vmax[v], __shfl_xor(vmax[v], 1, 32));
      vmax[v] = fmaxf(vmax[v], __shfl_xor(vmax[v], 2, 32));
      vmax[v] = fmaxf(vmax[v], __shfl_xor(vmax[v], 4, 32));
      vmax[v] = fmaxf(vmax[v], __shfl_xor(vmax[v], 8, 32));
    }
    if (lm == 0) {
      const int rb = ot * 16 + lh * 8;
#pragma unroll
      for (int v = 0; v < 8; ++v)
        outfeat[((size_t)b * 128 + rb + v) * MCTR + m] = vmax[v] * scale;
    }
  }
}

extern "C" void kernel_launch(void* const* d_in, const int* in_sizes, int n_in,
                              void* d_out, int out_size, void* d_ws, size_t ws_size,
                              hipStream_t stream) {
  const float* xyz      = (const float*)d_in[0];
  const float* features = (const float*)d_in[1];
  const float* W1 = (const float*)d_in[2];
  const float* b1 = (const float*)d_in[3];
  const float* W2 = (const float*)d_in[4];
  const float* b2 = (const float*)d_in[5];
  const float* W3 = (const float*)d_in[6];
  const float* b3 = (const float*)d_in[7];

  float* newxyz  = (float*)d_out;                       // (B, M, 3)
  float* outfeat = newxyz + (size_t)BATCH * MCTR * 3;   // (B, 128, M)

  int* gidx = (int*)d_ws;                               // (B*M, 64)
  int* gcnt = gidx + (size_t)BATCH * MCTR * NSAMP;      // (B*M)

  const size_t fps_smem = (size_t)4 * NPTS * sizeof(float) + 32 * sizeof(float) + 33 * sizeof(int);
  fps_kernel<<<BATCH, 1024, fps_smem, stream>>>(xyz, newxyz);

  ballq_kernel<<<(BATCH * MCTR * 32) / 256, 256, 0, stream>>>(xyz, newxyz, gidx, gcnt);

  const size_t mlp_smem = (size_t)(64 * W1S + 64 * W2S + 128 * W2S) * sizeof(_Float16)
                        + 256 * sizeof(float)
                        + (size_t)4 * PERWAVE_H * sizeof(_Float16);
  mlp_kernel<<<(BATCH * MCTR) / 4, 128, mlp_smem, stream>>>(
      xyz, features, W1, b1, W2, b2, W3, b3, newxyz, gidx, gcnt, outfeat);
}
